// MultiHeadSlotAttentionImplicit_4355096838596
// MI455X (gfx1250) — matrix operand order repaired and checked by reading, not compiled, not run on hardware
//
#include <hip/hip_runtime.h>

#define Bb 32
#define Nn 4096
#define Kk 8
#define Dd 256
#define Hh 4
#define DHd 64
#define NSPLIT 16
#define SCALE_ 0.125f
#define EPS_ 1e-8f

typedef __attribute__((ext_vector_type(16))) __bf16 v16bf;
typedef __attribute__((ext_vector_type(8)))  float  v8f;

union V16 { v16bf v; uint4 q[2]; unsigned short us[16]; };

__device__ __forceinline__ unsigned short f2bf(float f) {
  unsigned int u = __float_as_uint(f);
  u += 0x7FFFu + ((u >> 16) & 1u);
  return (unsigned short)(u >> 16);
}

__device__ __forceinline__ float wred32(float v) {
  #pragma unroll
  for (int o = 16; o > 0; o >>= 1) v += __shfl_xor(v, o, 32);
  return v;
}

// ---------------- slots = mu + exp(log_sigma) * noise ----------------
__global__ __launch_bounds__(256) void k_init_slots(
    const float* __restrict__ mu, const float* __restrict__ ls,
    const float* __restrict__ noise, float* __restrict__ slots) {
  int idx = blockIdx.x * 256 + threadIdx.x;            // B*K*D = 65536
  int d = idx & (Dd - 1);
  slots[idx] = mu[d] + __expf(ls[d]) * noise[idx];
}

// ---------------- fp32 -> bf16 weight conversion ----------------
__global__ __launch_bounds__(256) void k_prep_w(
    const float* __restrict__ Wk, const float* __restrict__ Wv,
    unsigned short* __restrict__ wkb, unsigned short* __restrict__ wvb) {
  int idx = blockIdx.x * 256 + threadIdx.x;            // 65536
  wkb[idx] = f2bf(Wk[idx]);
  wvb[idx] = f2bf(Wv[idx]);
}

// ---------------- LN(x) then K/V projection via WMMA bf16 ----------------
// block = 256 threads (8 waves), 16 token rows per block
// kf layout: [b][h][n][dh]   (dots B-matrix friendly)
// vt layout: [b][h][dh][n]   (attn@V B-matrix friendly)
__global__ __launch_bounds__(256) void k_proj(
    const float* __restrict__ x, const float* __restrict__ nw, const float* __restrict__ nb,
    const unsigned short* __restrict__ wkb, const unsigned short* __restrict__ wvb,
    unsigned short* __restrict__ kf, unsigned short* __restrict__ vt) {
  int t = threadIdx.x, wave = t >> 5, lane = t & 31;
  int half = lane >> 4, l16 = lane & 15;
  int r0 = blockIdx.x * 16;
  int b = r0 >> 12, n0 = r0 & (Nn - 1);
  __shared__ __align__(16) unsigned short xn_s[16 * 256];

  // layernorm 2 rows per wave -> bf16 in LDS
  #pragma unroll
  for (int rr = 0; rr < 2; rr++) {
    int m = wave * 2 + rr;
    const float* xrow = x + (size_t)(r0 + m) * Dd;
    float xv[8], s = 0.f, sq = 0.f;
    #pragma unroll
    for (int j = 0; j < 8; j++) { xv[j] = xrow[lane + 32 * j]; s += xv[j]; sq += xv[j] * xv[j]; }
    s = wred32(s); sq = wred32(sq);
    float mean = s * (1.f / 256.f);
    float var = sq * (1.f / 256.f) - mean * mean;
    float rstd = rsqrtf(var + 1e-5f);
    #pragma unroll
    for (int j = 0; j < 8; j++) {
      int col = lane + 32 * j;
      xn_s[m * 256 + col] = f2bf((xv[j] - mean) * rstd * nw[col] + nb[col]);
    }
  }
  __syncthreads();

  // 32 output tiles (16 for K, 16 for V), 4 per wave
  #pragma unroll
  for (int ti = 0; ti < 4; ti++) {
    int tt = wave + ti * 8;
    int proj = tt >> 4, jt = tt & 15, j0 = jt * 16;
    const unsigned short* wb = proj ? wvb : wkb;
    const unsigned short* wrow = wb + (size_t)(j0 + l16) * Dd;
    v8f c = {};
    #pragma unroll
    for (int d0 = 0; d0 < 256; d0 += 32) {
      V16 a, bm;
      const unsigned short* ar = xn_s + l16 * 256 + d0 + half * 8;
      a.q[0] = *(const uint4*)ar;
      a.q[1] = *(const uint4*)(ar + 16);
      const unsigned short* br = wrow + d0 + 8 * half;
      bm.q[0] = *(const uint4*)br;
      bm.q[1] = *(const uint4*)(br + 16);
      c = __builtin_amdgcn_wmma_f32_16x16x32_bf16(false, a.v, false, bm.v, (short)0, c, false, false);
    }
    int j = j0 + l16, h = j >> 6, dh = j & 63;
    if (proj == 0) {
      #pragma unroll
      for (int r = 0; r < 8; r++) {
        int n = n0 + r + 8 * half;
        kf[((size_t)(b * Hh + h) * Nn + n) * DHd + dh] = f2bf(c[r]);
      }
    } else {
      union { unsigned short us[8]; uint4 q; } pk;
      #pragma unroll
      for (int r = 0; r < 8; r++) pk.us[r] = f2bf(c[r]);
      *(uint4*)(vt + ((size_t)(b * Hh + h) * DHd + dh) * Nn + n0 + 8 * half) = pk.q;
    }
  }
}

// ---------------- q = LN(slots) @ Wq.T, bf16, padded to 16 slot rows ----------------
__global__ __launch_bounds__(256) void k_q(
    const float* __restrict__ slots, const float* __restrict__ nw, const float* __restrict__ nb,
    const float* __restrict__ Wq, unsigned short* __restrict__ qpad) {
  int b = blockIdx.x, t = threadIdx.x, wave = t >> 5, lane = t & 31;
  __shared__ float sn[8 * 256];
  {
    int k = wave;  // 8 waves, 8 slot rows
    const float* srow = slots + (size_t)(b * Kk + k) * Dd;
    float xv[8], s = 0.f, sq = 0.f;
    #pragma unroll
    for (int j = 0; j < 8; j++) { xv[j] = srow[lane + 32 * j]; s += xv[j]; sq += xv[j] * xv[j]; }
    s = wred32(s); sq = wred32(sq);
    float mean = s * (1.f / 256.f);
    float var = sq * (1.f / 256.f) - mean * mean;
    float rstd = rsqrtf(var + 1e-5f);
    #pragma unroll
    for (int j = 0; j < 8; j++) {
      int col = lane + 32 * j;
      sn[k * 256 + col] = (xv[j] - mean) * rstd * nw[col] + nb[col];
    }
  }
  __syncthreads();
  float acc[8] = {0, 0, 0, 0, 0, 0, 0, 0};
  const float* wrow = Wq + (size_t)t * Dd;
  for (int d = 0; d < 256; d++) {
    float wv = wrow[d];
    #pragma unroll
    for (int k = 0; k < 8; k++) acc[k] += sn[k * 256 + d] * wv;
  }
  int h = t >> 6, dh = t & 63;
  #pragma unroll
  for (int k = 0; k < 8; k++)
    qpad[((size_t)(b * Hh + h) * 16 + k) * DHd + dh] = f2bf(acc[k]);
  #pragma unroll
  for (int k = 8; k < 16; k++)
    qpad[((size_t)(b * Hh + h) * 16 + k) * DHd + dh] = 0;
}

// ---------------- fused attention: dots -> softmax(slots) -> attn@V (WMMA) ----------------
// one wave per block; grid = B*H*NSPLIT; each block handles 256 keys
__global__ __launch_bounds__(32) void k_attn(
    const unsigned short* __restrict__ qpad, const unsigned short* __restrict__ kf,
    const unsigned short* __restrict__ vt, float* __restrict__ upd_un,
    float* __restrict__ denom) {
  int lane = threadIdx.x, half = lane >> 4, l16 = lane & 15;
  int bid = blockIdx.x;
  int s = bid % NSPLIT, h = (bid / NSPLIT) % Hh, b = bid / (NSPLIT * Hh);
  int n0 = s * (Nn / NSPLIT);
  __shared__ __align__(16) unsigned short attn_s[16 * 32];
  #pragma unroll
  for (int i = lane; i < 16 * 32; i += 32) attn_s[i] = 0;  // rows 8..15 stay zero (pad slots)

  // Q as two A fragments (dh 0..31, 32..63)
  const unsigned short* qrow = qpad + ((size_t)(b * Hh + h) * 16 + l16) * DHd;
  V16 a0, a1;
  a0.q[0] = *(const uint4*)(qrow + 0 + half * 8);
  a0.q[1] = *(const uint4*)(qrow + 16 + half * 8);
  a1.q[0] = *(const uint4*)(qrow + 32 + half * 8);
  a1.q[1] = *(const uint4*)(qrow + 48 + half * 8);

  v8f cacc0 = {}, cacc1 = {}, cacc2 = {}, cacc3 = {};
  float dacc[8] = {0, 0, 0, 0, 0, 0, 0, 0};
  const unsigned short* kfb = kf + (size_t)(b * Hh + h) * Nn * DHd;
  const unsigned short* vtb = vt + (size_t)(b * Hh + h) * DHd * Nn;

  for (int nc = n0; nc < n0 + (Nn / NSPLIT); nc += 32) {
    #pragma unroll
    for (int tl = 0; tl < 2; tl++) {
      int n = nc + tl * 16 + l16;
      const unsigned short* krow = kfb + (size_t)n * DHd;
      V16 b0, b1;
      b0.q[0] = *(const uint4*)(krow + 0 + 8 * half);
      b0.q[1] = *(const uint4*)(krow + 16 + 8 * half);
      b1.q[0] = *(const uint4*)(krow + 32 + 8 * half);
      b1.q[1] = *(const uint4*)(krow + 48 + 8 * half);
      v8f c = {};
      c = __builtin_amdgcn_wmma_f32_16x16x32_bf16(false, a0.v, false, b0.v, (short)0, c, false, false);
      c = __builtin_amdgcn_wmma_f32_16x16x32_bf16(false, a1.v, false, b1.v, (short)0, c, false, false);
      float av[8];
      #pragma unroll
      for (int r = 0; r < 8; r++) av[r] = c[r] * SCALE_;
      if (half == 0) {  // lanes 0-15 hold the 8 real slots for key n -> in-lane softmax
        float mx = av[0];
        #pragma unroll
        for (int r = 1; r < 8; r++) mx = fmaxf(mx, av[r]);
        float sum = 0.f;
        #pragma unroll
        for (int r = 0; r < 8; r++) { av[r] = __expf(av[r] - mx); sum += av[r]; }
        float inv = 1.f / sum;
        #pragma unroll
        for (int r = 0; r < 8; r++) {
          float a = av[r] * inv;
          dacc[r] += a;
          attn_s[r * 32 + tl * 16 + l16] = f2bf(a);
        }
      }
    }
    asm volatile("s_wait_dscnt 0" ::: "memory");
    // attn as A fragment [16 slots x 32 keys]
    V16 aa;
    const unsigned short* arow = attn_s + l16 * 32;
    aa.q[0] = *(const uint4*)(arow + half * 8);
    aa.q[1] = *(const uint4*)(arow + 16 + half * 8);
    // 4 dh tiles of V
    {
      const unsigned short* vr0 = vtb + (size_t)(0 * 16 + l16) * Nn + nc + 8 * half;
      const unsigned short* vr1 = vtb + (size_t)(1 * 16 + l16) * Nn + nc + 8 * half;
      const unsigned short* vr2 = vtb + (size_t)(2 * 16 + l16) * Nn + nc + 8 * half;
      const unsigned short* vr3 = vtb + (size_t)(3 * 16 + l16) * Nn + nc + 8 * half;
      V16 bb;
      bb.q[0] = *(const uint4*)vr0; bb.q[1] = *(const uint4*)(vr0 + 16);
      cacc0 = __builtin_amdgcn_wmma_f32_16x16x32_bf16(false, aa.v, false, bb.v, (short)0, cacc0, false, false);
      bb.q[0] = *(const uint4*)vr1; bb.q[1] = *(const uint4*)(vr1 + 16);
      cacc1 = __builtin_amdgcn_wmma_f32_16x16x32_bf16(false, aa.v, false, bb.v, (short)0, cacc1, false, false);
      bb.q[0] = *(const uint4*)vr2; bb.q[1] = *(const uint4*)(vr2 + 16);
      cacc2 = __builtin_amdgcn_wmma_f32_16x16x32_bf16(false, aa.v, false, bb.v, (short)0, cacc2, false, false);
      bb.q[0] = *(const uint4*)vr3; bb.q[1] = *(const uint4*)(vr3 + 16);
      cacc3 = __builtin_amdgcn_wmma_f32_16x16x32_bf16(false, aa.v, false, bb.v, (short)0, cacc3, false, false);
    }
  }
  // denominator: sum over keys (lanes); lanes 16-31 contributed zero
  #pragma unroll
  for (int r = 0; r < 8; r++) dacc[r] = wred32(dacc[r]);
  if (lane == 0) {
    #pragma unroll
    for (int r = 0; r < 8; r++)
      atomicAdd(&denom[(size_t)(b * Hh + h) * 16 + r], dacc[r]);
  }
  // partial unnormalized update (slot rows 8..15 accumulate exact zeros)
  float* ub = upd_un + (size_t)(b * Hh + h) * 16 * DHd;
  #pragma unroll
  for (int r = 0; r < 8; r++) {
    int slot = r + 8 * half;
    atomicAdd(&ub[slot * DHd + 0 * 16 + l16], cacc0[r]);
    atomicAdd(&ub[slot * DHd + 1 * 16 + l16], cacc1[r]);
    atomicAdd(&ub[slot * DHd + 2 * 16 + l16], cacc2[r]);
    atomicAdd(&ub[slot * DHd + 3 * 16 + l16], cacc3[r]);
  }
}

// ---------------- upd_m[b,k,(h,dh)] = upd_un / (denom + eps) ----------------
__global__ __launch_bounds__(256) void k_finalize(
    const float* __restrict__ upd_un, const float* __restrict__ denom,
    float* __restrict__ upd_m) {
  int idx = blockIdx.x * 256 + threadIdx.x;            // B*K*D
  int b = idx >> 11;               // / (K*D) = /2048
  int rem = idx & 2047;
  int k = rem >> 8, j = rem & 255;
  int h = j >> 6, dh = j & 63;
  float den = denom[(size_t)(b * Hh + h) * 16 + k] + EPS_;
  upd_m[idx] = upd_un[((size_t)(b * Hh + h) * 16 + k) * DHd + dh] / den;
}

// ---------------- generic 256x256 row-GEMM: out = [relu](in@W.T + bias) [+ res] ----------------
__global__ __launch_bounds__(256) void k_rowgemm(
    const float* __restrict__ in, const float* __restrict__ W,
    const float* __restrict__ bias, const float* __restrict__ res,
    float* __restrict__ out, int relu) {
  int r = blockIdx.x, t = threadIdx.x;
  __shared__ float row[256];
  row[t] = in[(size_t)r * 256 + t];
  __syncthreads();
  float acc = bias ? bias[t] : 0.f;
  const float* wr = W + (size_t)t * 256;
  for (int d = 0; d < 256; d++) acc += row[d] * wr[d];
  if (relu) acc = fmaxf(acc, 0.f);
  if (res) acc += res[(size_t)r * 256 + t];
  out[(size_t)r * 256 + t] = acc;
}

// ---------------- fused GRU cell: slots = GRU(updo, slots) ----------------
__global__ __launch_bounds__(256) void k_gru(
    const float* __restrict__ updo, float* __restrict__ slots,
    const float* __restrict__ Wih, const float* __restrict__ Whh,
    const float* __restrict__ bih, const float* __restrict__ bhh) {
  int r = blockIdx.x, t = threadIdx.x;
  __shared__ float u[256], sfull[256];
  u[t] = updo[(size_t)r * 256 + t];
  sfull[t] = slots[(size_t)r * 256 + t];
  __syncthreads();
  float ir = bih[t], iz = bih[256 + t], inn = bih[512 + t];
  float hr = bhh[t], hz = bhh[256 + t], hn = bhh[512 + t];
  const float* wi0 = Wih + (size_t)t * 256;
  const float* wi1 = Wih + (size_t)(256 + t) * 256;
  const float* wi2 = Wih + (size_t)(512 + t) * 256;
  const float* wh0 = Whh + (size_t)t * 256;
  const float* wh1 = Whh + (size_t)(256 + t) * 256;
  const float* wh2 = Whh + (size_t)(512 + t) * 256;
  for (int d = 0; d < 256; d++) {
    float ud = u[d], sd = sfull[d];
    ir += ud * wi0[d]; iz += ud * wi1[d]; inn += ud * wi2[d];
    hr += sd * wh0[d]; hz += sd * wh1[d]; hn += sd * wh2[d];
  }
  float rr = 1.f / (1.f + __expf(-(ir + hr)));
  float zz = 1.f / (1.f + __expf(-(iz + hz)));
  float nn = tanhf(inn + rr * hn);
  slots[(size_t)r * 256 + t] = (1.f - zz) * nn + zz * sfull[t];
}

// ---------------- row layernorm (for MLP branch) ----------------
__global__ __launch_bounds__(256) void k_ln_row(
    const float* __restrict__ in, const float* __restrict__ nw, const float* __restrict__ nb,
    float* __restrict__ out) {
  int r = blockIdx.x, t = threadIdx.x, wave = t >> 5, lane = t & 31;
  __shared__ float red[16];
  float xv = in[(size_t)r * 256 + t];
  float s = wred32(xv), sq = wred32(xv * xv);
  if (lane == 0) { red[wave] = s; red[8 + wave] = sq; }
  __syncthreads();
  float tot = 0.f, totq = 0.f;
  #pragma unroll
  for (int i = 0; i < 8; i++) { tot += red[i]; totq += red[8 + i]; }
  float mean = tot * (1.f / 256.f);
  float var = totq * (1.f / 256.f) - mean * mean;
  out[(size_t)r * 256 + t] = (xv - mean) * rsqrtf(var + 1e-5f) * nw[t] + nb[t];
}

extern "C" void kernel_launch(void* const* d_in, const int* in_sizes, int n_in,
                              void* d_out, int out_size, void* d_ws, size_t ws_size,
                              hipStream_t stream) {
  const float* x      = (const float*)d_in[0];
  const float* noise  = (const float*)d_in[1];
  const float* mu     = (const float*)d_in[2];
  const float* lsig   = (const float*)d_in[3];
  const float* nin_w  = (const float*)d_in[4];
  const float* nin_b  = (const float*)d_in[5];
  const float* nsl_w  = (const float*)d_in[6];
  const float* nsl_b  = (const float*)d_in[7];
  const float* nml_w  = (const float*)d_in[8];
  const float* nml_b  = (const float*)d_in[9];
  const float* Wq     = (const float*)d_in[10];
  const float* Wk     = (const float*)d_in[11];
  const float* Wv     = (const float*)d_in[12];
  const float* Wo     = (const float*)d_in[13];
  const float* Wih    = (const float*)d_in[14];
  const float* Whh    = (const float*)d_in[15];
  const float* bih    = (const float*)d_in[16];
  const float* bhh    = (const float*)d_in[17];
  const float* W1     = (const float*)d_in[18];
  const float* b1     = (const float*)d_in[19];
  const float* W2     = (const float*)d_in[20];
  const float* b2     = (const float*)d_in[21];
  float* slots = (float*)d_out;                        // [B,K,D] fp32 = output

  char* wp = (char*)d_ws;
  auto carve = [&](size_t bytes) { char* p = wp; wp += (bytes + 255) & ~(size_t)255; return p; };
  unsigned short* wkb    = (unsigned short*)carve((size_t)Dd * Dd * 2);
  unsigned short* wvb    = (unsigned short*)carve((size_t)Dd * Dd * 2);
  unsigned short* kf     = (unsigned short*)carve((size_t)Bb * Hh * Nn * DHd * 2);  // 64 MB
  unsigned short* vt     = (unsigned short*)carve((size_t)Bb * Hh * DHd * Nn * 2);  // 64 MB
  unsigned short* qpad   = (unsigned short*)carve((size_t)Bb * Hh * 16 * DHd * 2);
  float* upd_un          = (float*)carve((size_t)Bb * Hh * 16 * DHd * 4);
  float* denom           = (float*)carve((size_t)Bb * Hh * 16 * 4);
  float* upd_m           = (float*)carve((size_t)Bb * Kk * Dd * 4);
  float* updo            = (float*)carve((size_t)Bb * Kk * Dd * 4);
  float* slots_ln        = (float*)carve((size_t)Bb * Kk * Dd * 4);
  float* h1              = (float*)carve((size_t)Bb * Kk * Dd * 4);

  k_prep_w<<<256, 256, 0, stream>>>(Wk, Wv, wkb, wvb);
  k_init_slots<<<256, 256, 0, stream>>>(mu, lsig, noise, slots);
  k_proj<<<(Bb * Nn) / 16, 256, 0, stream>>>(x, nin_w, nin_b, wkb, wvb, kf, vt);

  for (int step = 0; step < 3; step++) {
    k_q<<<Bb, 256, 0, stream>>>(slots, nsl_w, nsl_b, Wq, qpad);
    hipMemsetAsync(upd_un, 0, (size_t)Bb * Hh * 16 * DHd * 4, stream);
    hipMemsetAsync(denom, 0, (size_t)Bb * Hh * 16 * 4, stream);
    k_attn<<<Bb * Hh * NSPLIT, 32, 0, stream>>>(qpad, kf, vt, upd_un, denom);
    k_finalize<<<(Bb * Kk * Dd) / 256, 256, 0, stream>>>(upd_un, denom, upd_m);
    k_rowgemm<<<Bb * Kk, 256, 0, stream>>>(upd_m, Wo, nullptr, nullptr, updo, 0);
    k_gru<<<Bb * Kk, 256, 0, stream>>>(updo, slots, Wih, Whh, bih, bhh);
    k_ln_row<<<Bb * Kk, 256, 0, stream>>>(slots, nml_w, nml_b, slots_ln);
    k_rowgemm<<<Bb * Kk, 256, 0, stream>>>(slots_ln, W1, b1, nullptr, h1, 1);
    k_rowgemm<<<Bb * Kk, 256, 0, stream>>>(h1, W2, b2, slots, slots, 0);
  }
}
